// YoloLoss_50757923504767
// MI455X (gfx1250) — compile-verified, matrix operand order checked by reading, not run on hardware
//
#include <hip/hip_runtime.h>
#include <math.h>

#define B_    128
#define W_    19
#define H_    19
#define A_    5
#define C_    81
#define M_    24
#define N_    (W_*H_)       // 361
#define NA_   (N_*A_)       // 1805
#define BNA_  (B_*NA_)      // 231040
#define TILES_ (BNA_/16)    // 14440 (exact)
#define EPS_  1e-7f
#define NBUCK 256
// bucket terms: 0 noobj, 1 obj, 2 xy, 3 wh, 4 score

typedef __attribute__((ext_vector_type(16))) _Float16 v16h;
typedef __attribute__((ext_vector_type(8)))  float    v8f;

__device__ __forceinline__ double waveRedAdd(double v) {
#pragma unroll
  for (int off = 16; off > 0; off >>= 1) v += __shfl_down(v, off, 32);
  return v;
}

// ---------------- init: zero buckets/gtsum, tclass := background (C-1) -------
__global__ void kInit(double* bucket, float* gtsum, int* tclass) {
  int t = blockIdx.x * blockDim.x + threadIdx.x;
  if (t < 5 * NBUCK) bucket[t] = 0.0;
  if (t < NA_) gtsum[t] = 0.0f;
  if (t < BNA_) tclass[t] = C_ - 1;
}

// ---------------- assignment: one thread per (b, m) gt box -------------------
__global__ void kAssign(const float* __restrict__ conf,
                        const float* __restrict__ pred_xy,
                        const float* __restrict__ pred_wh,
                        const float* __restrict__ true_label,
                        const float* __restrict__ true_object,
                        double* bucket, float* gtsum, int* tclass) {
  int idx = blockIdx.x * blockDim.x + threadIdx.x;
  if (idx >= B_ * M_) return;
  const int b = idx / M_;
  const float aw[A_] = {0.57273f, 1.87446f, 3.33843f, 7.88282f, 9.77052f};
  const float ah[A_] = {0.677385f, 2.06253f, 5.47434f, 3.52778f, 9.16828f};

  const float* to = true_object + (size_t)idx * 4;
  const float inv = 1.0f / 32.0f;
  float gx = to[0] * inv, gy = to[1] * inv, gw = to[2] * inv, gh = to[3] * inv;
  int ci = (int)gy, cj = (int)gx;
  int cell = ci * H_ + cj;
  float ga = gw * gh;

  int best = 0; float bi = -1.0f; bool over[A_];
#pragma unroll
  for (int a = 0; a < A_; ++a) {
    float inter = fminf(gw, aw[a]) * fminf(gh, ah[a]);
    float iou = inter / (ga + aw[a] * ah[a] - inter + EPS_);
    over[a] = iou > 0.6f;
    if (iou > bi) { bi = iou; best = a; }   // first-max == jnp.argmax
  }

  // label = argmax of one-hot row
  const float* tl = true_label + (size_t)idx * C_;
  int label = 0; float lb = tl[0];
  for (int c = 1; c < C_; ++c) { float v = tl[c]; if (v > lb) { lb = v; label = c; } }

  size_t base = (size_t)b * NA_ + (size_t)cell * A_;
  // gathers at the best anchor
  float pw = pred_wh[(base + best) * 2 + 0];
  float ph = pred_wh[(base + best) * 2 + 1];
  float inter = fminf(pw, gw) * fminf(ph, gh);
  float ioupg = inter / (pw * ph + ga - inter + EPS_);
  float sw = 2.0f - (gw * (1.0f / (float)W_)) * (gh * (1.0f / (float)W_));
  float px = pred_xy[(base + best) * 2 + 0];
  float py = pred_xy[(base + best) * 2 + 1];
  float dx = px - gx, dy = py - gy, dw = pw - gw, dh = ph - gh;  // cord cancels
  double xyv = (double)(sw * (dx * dx + dy * dy));
  double whv = (double)(sw * (dw * dw + dh * dh));

  double noc = 0.0, objv = 0.0;
#pragma unroll
  for (int a = 0; a < A_; ++a) {
    bool mb = (a == best);
    if (mb || over[a]) {
      float cp = fminf(fmaxf(conf[base + a], EPS_), 1.0f - EPS_);
      noc += (double)__logf(1.0f - cp);          // removes this pos from noobj sum
      tclass[base + a] = label;
      if (mb) objv = -(double)(ioupg * __logf(cp) + (1.0f - ioupg) * __logf(1.0f - cp));
    }
  }
  atomicAdd(&gtsum[cell * A_ + best], 1.0f);
  int bk = idx & (NBUCK - 1);
  atomicAdd(&bucket[0 * NBUCK + bk], noc);
  atomicAdd(&bucket[1 * NBUCK + bk], objv);
  atomicAdd(&bucket[2 * NBUCK + bk], xyv);
  atomicAdd(&bucket[3 * NBUCK + bk], whv);
}

// ---------------- noobj base: sum -log(1-cp) over ALL positions --------------
__global__ void kNoObj(const float* __restrict__ conf, double* bucket) {
  int idx = blockIdx.x * blockDim.x + threadIdx.x;
  double v = 0.0;
  if (idx < BNA_) {
    float cp = fminf(fmaxf(conf[idx], EPS_), 1.0f - EPS_);
    v = -(double)__logf(1.0f - cp);
  }
  v = waveRedAdd(v);
  if ((threadIdx.x & 31) == 0) {
    int wid = blockIdx.x * (blockDim.x >> 5) + (threadIdx.x >> 5);
    atomicAdd(&bucket[0 * NBUCK + (wid & (NBUCK - 1))], v);
  }
}

// ---------------- CE stream: 16 rows/wave, row-sums of exp via WMMA ----------
__global__ void kCE(const float* __restrict__ cls,
                    const float* __restrict__ gtsum,
                    const int* __restrict__ tclass,
                    double* bucket) {
  int lane = threadIdx.x & 31;
  int wid = blockIdx.x * (blockDim.x >> 5) + (threadIdx.x >> 5);  // tile id
  if (wid >= TILES_) return;
  int rowBase = wid * 16;
  int r = lane & 15, hi = lane >> 4;
  const float* rp = cls + (size_t)(rowBase + r) * C_;

  v16h ones;
#pragma unroll
  for (int t = 0; t < 16; ++t) ones[t] = (_Float16)1.0f;
  v8f acc;
#pragma unroll
  for (int t = 0; t < 8; ++t) acc[t] = 0.0f;

  // A fragment (f16 16x32): lanes 0-15 row M=lane, elems 0..7 -> K, 8..15 -> K+16;
  // hi lanes shift K by 8. 3 chunks cover K=0..95 (cols >= 81 padded with exp=0).
#pragma unroll
  for (int c = 0; c < 3; ++c) {
    v16h a;
    int k0 = 32 * c + 8 * hi;
#pragma unroll
    for (int t = 0; t < 8; ++t) {
      int k = k0 + t;
      float e0 = 0.0f, e1 = 0.0f;
      if (k < C_)      e0 = __expf(rp[k]);
      if (k + 16 < C_) e1 = __expf(rp[k + 16]);
      a[t] = (_Float16)e0;
      a[t + 8] = (_Float16)e1;
    }
    acc = __builtin_amdgcn_wmma_f32_16x16x32_f16(false, a, false, ones,
                                                 (short)0, acc, false, false);
  }

  // D: lanes 0-15 vgpr j -> row j; lanes 16-31 vgpr j -> row 8+j (all cols equal)
  float rs = 0.0f;
#pragma unroll
  for (int j = 0; j < 8; ++j)
    if (lane == j || lane == j + 16) rs = acc[j];

  double part = 0.0;
  bool owner = (lane < 8) || (lane >= 16 && lane < 24);
  if (owner) {
    int m = (lane < 8) ? lane : (lane - 8);
    int ri = rowBase + m;
    int na = ri % NA_;
    float g = gtsum[na];
    if (g != 0.0f) {
      float csl = cls[(size_t)ri * C_ + tclass[ri]];
      part = (double)(g * (__logf(rs) - csl));   // gt-count * (lse - cs[label])
    }
  }
  part = waveRedAdd(part);
  if (lane == 0)
    atomicAdd(&bucket[4 * NBUCK + (wid & (NBUCK - 1))], part);
}

// ---------------- finalize: reduce buckets, scale, write 6 outputs -----------
__global__ void kFinal(const double* __restrict__ bucket, float* __restrict__ out) {
  int lane = threadIdx.x & 31;
  double v[5];
#pragma unroll
  for (int t = 0; t < 5; ++t) {
    double s = 0.0;
    for (int i = lane; i < NBUCK; i += 32) s += bucket[t * NBUCK + i];
    v[t] = waveRedAdd(s);
  }
  if (lane == 0) {
    double invB = 1.0 / (double)B_;
    double noobj = 1.0 * v[0] * invB;   // SCALE_NOOBJ
    double obj   = 5.0 * v[1] * invB;   // SCALE_OBJ
    double xy    = 5.0 * v[2] * invB;
    double wh    = 5.0 * v[3] * invB;
    double score = 5.0 * v[4] * invB;
    double total = noobj + obj + xy + wh + score;
    out[0] = (float)total; out[1] = (float)noobj; out[2] = (float)obj;
    out[3] = (float)score; out[4] = (float)xy;    out[5] = (float)wh;
  }
}

extern "C" void kernel_launch(void* const* d_in, const int* in_sizes, int n_in,
                              void* d_out, int out_size, void* d_ws, size_t ws_size,
                              hipStream_t stream) {
  const float* conf        = (const float*)d_in[1];
  const float* pred_xy     = (const float*)d_in[2];
  const float* pred_wh     = (const float*)d_in[3];
  const float* cls_score   = (const float*)d_in[4];
  const float* true_label  = (const float*)d_in[5];
  const float* true_object = (const float*)d_in[6];
  // d_in[0]=epoch (unused), d_in[7]=fm_cord (cancels analytically)

  double* bucket = (double*)d_ws;
  float*  gtsum  = (float*)((char*)d_ws + 5 * NBUCK * sizeof(double));
  int*    tclass = (int*)((char*)gtsum + ((NA_ + 3) & ~3) * sizeof(float));

  kInit<<<(BNA_ + 255) / 256, 256, 0, stream>>>(bucket, gtsum, tclass);
  kAssign<<<(B_ * M_ + 255) / 256, 256, 0, stream>>>(conf, pred_xy, pred_wh,
                                                     true_label, true_object,
                                                     bucket, gtsum, tclass);
  kNoObj<<<(BNA_ + 255) / 256, 256, 0, stream>>>(conf, bucket);
  kCE<<<TILES_ / 8, 256, 0, stream>>>(cls_score, gtsum, tclass, bucket);
  kFinal<<<1, 32, 0, stream>>>(bucket, (float*)d_out);
}
